// MPN_2D_58334245814980
// MI455X (gfx1250) — compile-verified, matrix operand order checked by reading, not run on hardware
//
#include <hip/hip_runtime.h>
#include <hip/hip_bf16.h>
#include <math.h>

// ---------------------------------------------------------------------------
// CMPNN forward for MI455X (gfx1250, wave32). All GEMMs run through a tiled
// v_wmma_f32_16x16x32_bf16 kernel (bf16 inputs / f32 accumulate), weights are
// packed to zero-padded bf16 once per launch. Gather / GRU-gate / reductions
// are fp32 scalar kernels.
// Block = 128 threads = 4 wave32s; 64x64 output tile; each wave computes a
// 32x32 sub-tile = 4 WMMAs per 32-wide K panel. All large GEMMs read
// zero-padded, 16B-aligned activations so the staging path is fully
// vectorized (global_load_b128 -> v_cvt_pk_bf16_f32 -> ds_store_b128).
// ---------------------------------------------------------------------------

typedef __bf16 bf16;
typedef __attribute__((ext_vector_type(16))) __bf16 v16bf;
typedef __attribute__((ext_vector_type(8)))  __bf16 v8bf;
typedef __attribute__((ext_vector_type(8)))  float  v8f;

#define HDIM 300          // hidden size
#define LD   320          // row stride for H-sized activations (16B aligned)
#define G3   900          // 3*H (GRU gates / concat width)
#define GP   928          // padded 3*H (multiple of 32)
#define NB   6            // max neighbors
#define AFD  133          // atom feature dim
#define BFD  147          // bond feature dim
#define FPAD 160          // padded feature dim (covers 133 and 147)
#define CATW 900          // concat(agg, message_atom, input_atom) width
#define MSG2 600          // 2*H GRU output width
#define APM  64           // atoms per molecule (fixed by reference)
#define LDK  40           // LDS K-panel row stride in bf16 (32 + 8 pad)

// ---------------------------------------------------------------------------
// Weight packing: [out,K] f32 row-major  ->  [outPad,Kpad] bf16, zero padded
// ---------------------------------------------------------------------------
__global__ void pack_weight_kernel(const float* __restrict__ W, bf16* __restrict__ Wp,
                                   int out, int K, int Kpad, int outPad) {
  int idx = blockIdx.x * blockDim.x + threadIdx.x;
  int total = outPad * Kpad;
  if (idx >= total) return;
  int r = idx / Kpad, k = idx - r * Kpad;
  float v = (r < out && k < K) ? W[(long)r * K + k] : 0.0f;
  Wp[idx] = (bf16)v;
}

// Repack activations: [rows,K] f32 (odd stride) -> [rows,Kpad] f32, zero padded
__global__ void pack_act_kernel(const float* __restrict__ src, float* __restrict__ dst,
                                int rows, int K, int Kpad) {
  long idx = (long)blockIdx.x * blockDim.x + threadIdx.x;
  long total = (long)rows * Kpad;
  if (idx >= total) return;
  int r = (int)(idx / Kpad), k = (int)(idx - (long)r * Kpad);
  dst[idx] = (k < K) ? src[(long)r * K + k] : 0.0f;
}

// ---------------------------------------------------------------------------
// Generic GEMM: C[rows,out] = actOut( actA(A)[rows,K] @ Wp[out,K]^T + bias + addend )
// Columns [outc, padTo) of C are zero-filled so C can later be consumed with
// K = row stride (keeps downstream GEMMs on the vectorized staging path).
// Fragment layout per CDNA5 ISA 7.12.2 (16-bit A 16x32 / B 32x16):
//   lanes 0-15: row = lane,    K in {0..7} u {16..23}
//   lanes16-31: row = lane-16, K in {8..15} u {24..31}
// ---------------------------------------------------------------------------
__global__ void __launch_bounds__(128)
gemm_wmma(const float* __restrict__ A, int lda, int rows, int K,
          const bf16* __restrict__ Wp, int Kpad, int outPad,
          const float* __restrict__ bias,
          const float* __restrict__ addend, int ldadd,
          float* __restrict__ C, int ldc, int outc, int padTo,
          int actA, int actOut)
{
  __shared__ bf16 As[64 * LDK];
  __shared__ bf16 Ws[64 * LDK];

  const int rowBase = blockIdx.y * 64;
  const int colBase = blockIdx.x * 64;
  const int tid  = threadIdx.x;
  const int wave = tid >> 5;
  const int lane = tid & 31;
  const int wr   = (wave >> 1) & 1;   // wave's 32-row half of the 64x64 tile
  const int wc   = wave & 1;          // wave's 32-col half
  const int half = lane >> 4;         // K-half selector per fragment layout
  const int rr   = lane & 15;         // row/col within 16
  const int srow = tid >> 2;          // staging: row 0..31 (and +32)
  const int scol = (tid & 3) * 8;     // staging: k offset 0/8/16/24

  // uniform: can we use aligned float4 loads for the whole row block?
  const bool vecRows = ((lda & 3) == 0) && (rowBase + 64 <= rows);

  // LDS fragment base pointers (constant across K panels)
  const bf16* apA0 = &As[(wr * 32 +  0 + rr) * LDK + half * 8];
  const bf16* apA1 = &As[(wr * 32 + 16 + rr) * LDK + half * 8];
  const bf16* bpB0 = &Ws[(wc * 32 +  0 + rr) * LDK + half * 8];
  const bf16* bpB1 = &Ws[(wc * 32 + 16 + rr) * LDK + half * 8];

  v8f acc00 = {0.f,0.f,0.f,0.f,0.f,0.f,0.f,0.f};
  v8f acc01 = acc00, acc10 = acc00, acc11 = acc00;

  // W staging source rows, clamped into the packed (zero-padded) weight array
  const int w0 = (colBase + srow      < outPad) ? (colBase + srow)      : (outPad - 1);
  const int w1 = (colBase + srow + 32 < outPad) ? (colBase + srow + 32) : (outPad - 1);

  for (int k0 = 0; k0 < Kpad; k0 += 32) {
    const bool fullK = (k0 + 32 <= K);

    // ---- stage A panel: f32 -> bf16, optional ReLU (branch-free) ----
    #pragma unroll
    for (int rep = 0; rep < 2; ++rep) {
      const int gr = rowBase + srow + rep * 32;
      v8bf pkt;
      if (vecRows && fullK) {
        const float4 lo = *(const float4*)&A[(long)gr * lda + k0 + scol];
        const float4 hi = *(const float4*)&A[(long)gr * lda + k0 + scol + 4];
        float f[8] = {lo.x, lo.y, lo.z, lo.w, hi.x, hi.y, hi.z, hi.w};
        #pragma unroll
        for (int i = 0; i < 8; ++i) {
          float v = f[i];
          if (actA) v = fmaxf(v, 0.0f);
          pkt[i] = (bf16)v;
        }
      } else {
        const int grc = (gr < rows) ? gr : (rows - 1);
        #pragma unroll
        for (int i = 0; i < 8; ++i) {
          const int gk  = k0 + scol + i;
          const int gkc = (gk < K) ? gk : 0;
          float v = A[(long)grc * lda + gkc];   // always in-bounds
          v = (gk < K) ? v : 0.0f;              // cndmask, no branch
          if (actA) v = fmaxf(v, 0.0f);
          pkt[i] = (bf16)v;
        }
      }
      *(v8bf*)&As[(srow + rep * 32) * LDK + scol] = pkt;
    }
    if (k0 + 32 < Kpad) {
      int grp = rowBase + srow;
      if (grp >= rows) grp = rows - 1;
      __builtin_prefetch(&A[(long)grp * lda + k0 + 32 + scol], 0, 0); // global_prefetch_b8
    }

    // ---- stage W panel: packed bf16, clamped rows are zero padding ----
    *(v8bf*)&Ws[srow * LDK + scol] =
        *(const v8bf*)&Wp[(long)w0 * Kpad + k0 + scol];
    *(v8bf*)&Ws[(srow + 32) * LDK + scol] =
        *(const v8bf*)&Wp[(long)w1 * Kpad + k0 + scol];

    __syncthreads();

    // ---- build fragments (two contiguous 16B LDS chunks each) + 4 WMMAs ----
    v16bf a0, a1, b0, b1;
    #pragma unroll
    for (int i = 0; i < 8; ++i) {
      a0[i] = apA0[i];  a0[8 + i] = apA0[16 + i];
      a1[i] = apA1[i];  a1[8 + i] = apA1[16 + i];
      b0[i] = bpB0[i];  b0[8 + i] = bpB0[16 + i];
      b1[i] = bpB1[i];  b1[8 + i] = bpB1[16 + i];
    }
    acc00 = __builtin_amdgcn_wmma_f32_16x16x32_bf16(false, a0, false, b0, (short)0, acc00, false, false);
    acc01 = __builtin_amdgcn_wmma_f32_16x16x32_bf16(false, a0, false, b1, (short)0, acc01, false, false);
    acc10 = __builtin_amdgcn_wmma_f32_16x16x32_bf16(false, a1, false, b0, (short)0, acc10, false, false);
    acc11 = __builtin_amdgcn_wmma_f32_16x16x32_bf16(false, a1, false, b1, (short)0, acc11, false, false);
    __syncthreads();
  }

  // ---- store: CDNA5 f32 C/D layout: VGPR v -> M = half*8 + v, col = lane&15 ----
  const v8f* accs[4] = {&acc00, &acc01, &acc10, &acc11};
  #pragma unroll
  for (int tr = 0; tr < 2; ++tr) {
    #pragma unroll
    for (int tc = 0; tc < 2; ++tc) {
      const v8f& acc = *accs[tr * 2 + tc];
      #pragma unroll
      for (int v = 0; v < 8; ++v) {
        const int gr = rowBase + wr * 32 + tr * 16 + half * 8 + v;
        const int gc = colBase + wc * 32 + tc * 16 + rr;
        if (gr < rows && gc < padTo) {
          float val = 0.0f;
          if (gc < outc) {
            val = acc[v];
            if (bias)   val += bias[gc];
            if (addend) val += addend[(long)gr * ldadd + gc];
            if (actOut) val = fmaxf(val, 0.0f);
          }
          C[(long)gr * ldc + gc] = val;   // zero-fills pad columns
        }
      }
    }
  }
}

// ---------------------------------------------------------------------------
// Elementwise / gather kernels
// ---------------------------------------------------------------------------
__global__ void agg_add_kernel(const float* __restrict__ mb, const int* __restrict__ a2b,
                               float* __restrict__ ma, int nrows) {
  int idx = blockIdx.x * blockDim.x + threadIdx.x;
  if (idx >= nrows * HDIM) return;
  int v = idx / HDIM, hh = idx - v * HDIM;
  float s = 0.0f, mx = -__builtin_inff();
  #pragma unroll
  for (int j = 0; j < NB; ++j) {
    int b = a2b[v * NB + j];
    float x = mb[(long)b * LD + hh];
    s += x; mx = fmaxf(mx, x);
  }
  ma[(long)v * LD + hh] += s * mx;
}

// tmp = message_atom[b2a] - message_bond[b2revb]; zero-fills pad cols 300..319
__global__ void bond_tmp_kernel(const float* __restrict__ ma, const float* __restrict__ mb,
                                const int* __restrict__ b2a, const int* __restrict__ b2revb,
                                float* __restrict__ tmp, int brows) {
  long idx = (long)blockIdx.x * blockDim.x + threadIdx.x;
  if (idx >= (long)brows * LD) return;
  int b = (int)(idx / LD), hh = (int)(idx - (long)b * LD);
  float val = 0.0f;
  if (hh < HDIM)
    val = ma[(long)b2a[b] * LD + hh] - mb[(long)b2revb[b] * LD + hh];
  tmp[idx] = val;
}

__global__ void build_node_in_kernel(const float* __restrict__ mb, const int* __restrict__ a2b,
                                     const float* __restrict__ ma, const float* __restrict__ ia,
                                     float* __restrict__ ni, int nrows) {
  int idx = blockIdx.x * blockDim.x + threadIdx.x;
  if (idx >= nrows * HDIM) return;
  int v = idx / HDIM, hh = idx - v * HDIM;
  float s = 0.0f, mx = -__builtin_inff();
  #pragma unroll
  for (int j = 0; j < NB; ++j) {
    int b = a2b[v * NB + j];
    float x = mb[(long)b * LD + hh];
    s += x; mx = fmaxf(mx, x);
  }
  ni[(long)v * CATW + hh]            = s * mx;
  ni[(long)v * CATW + HDIM + hh]     = ma[(long)v * LD + hh];
  ni[(long)v * CATW + 2 * HDIM + hh] = ia[(long)v * LD + hh];
}

// h0 = per-molecule max of pre-relu node; zero-fills pad cols 300..319
__global__ void h0_kernel(const float* __restrict__ node,
                          float* __restrict__ hf, float* __restrict__ hb, int M) {
  int idx = blockIdx.x * blockDim.x + threadIdx.x;
  if (idx >= M * LD) return;
  int m = idx / LD, hh = idx - m * LD;
  float mx = 0.0f;
  if (hh < HDIM) {
    mx = -__builtin_inff();
    for (int a = 0; a < APM; ++a)
      mx = fmaxf(mx, node[(long)(1 + m * APM + a) * LD + hh]);
  }
  hf[idx] = mx;
  hb[idx] = mx;
}

__global__ void row0_kernel(const float* __restrict__ node, float* __restrict__ msgout) {
  int hh = blockIdx.x * blockDim.x + threadIdx.x;
  if (hh >= HDIM) return;
  float v = fmaxf(node[hh], 0.0f);
  msgout[hh]        = v;
  msgout[HDIM + hh] = v;
}

// GRU gate combine (torch order r,z,n). gi rows are (m*APM + timestep) with GP stride.
__global__ void gru_gate_kernel(const float* __restrict__ gi, const float* __restrict__ gh,
                                float* __restrict__ h, float* __restrict__ outmsg,
                                int M, int t, int dir) {
  int idx = blockIdx.x * blockDim.x + threadIdx.x;
  if (idx >= M * HDIM) return;
  int m = idx / HDIM, hh = idx - m * HDIM;
  int ts = dir ? (APM - 1 - t) : t;
  long xrow = (long)m * APM + ts;
  float ir  = gi[xrow * GP + hh];
  float iz  = gi[xrow * GP + HDIM + hh];
  float inn = gi[xrow * GP + 2 * HDIM + hh];
  float hr  = gh[(long)m * GP + hh];
  float hz  = gh[(long)m * GP + HDIM + hh];
  float hn  = gh[(long)m * GP + 2 * HDIM + hh];
  float r = 1.0f / (1.0f + expf(-(ir + hr)));
  float z = 1.0f / (1.0f + expf(-(iz + hz)));
  float n = tanhf(inn + r * hn);
  float hp = h[(long)m * LD + hh];
  float hnew = (1.0f - z) * n + z * hp;
  h[(long)m * LD + hh] = hnew;
  outmsg[(long)(1 + xrow) * MSG2 + (dir ? HDIM : 0) + hh] = hnew;
}

__global__ void mean_kernel(const float* __restrict__ ah, float* __restrict__ out, int M) {
  int idx = blockIdx.x * blockDim.x + threadIdx.x;
  if (idx >= M * HDIM) return;
  int m = idx / HDIM, hh = idx - m * HDIM;
  float s = 0.0f;
  for (int a = 0; a < APM; ++a)
    s += ah[(long)(1 + m * APM + a) * LD + hh];
  out[(long)m * HDIM + hh] = s * (1.0f / (float)APM);
}

// ---------------------------------------------------------------------------
// Host-side launcher
// ---------------------------------------------------------------------------
static inline void launch_gemm(const float* A, int lda, int rows, int K,
                               const bf16* Wp, int Kpad, const float* bias,
                               const float* addend, int ldadd,
                               float* C, int ldc, int outc, int padTo,
                               int actA, int actOut, hipStream_t s) {
  int outPad = ((outc + 31) / 32) * 32;
  dim3 grid((outc + 63) / 64, (rows + 63) / 64);
  gemm_wmma<<<grid, 128, 0, s>>>(A, lda, rows, K, Wp, Kpad, outPad, bias, addend,
                                 ldadd, C, ldc, outc, padTo, actA, actOut);
}

static inline void launch_pack(const float* W, bf16* Wp, int out, int K, hipStream_t s) {
  int outPad = ((out + 31) / 32) * 32;
  int Kpad   = ((K   + 31) / 32) * 32;
  int total  = outPad * Kpad;
  pack_weight_kernel<<<(total + 255) / 256, 256, 0, s>>>(W, Wp, out, K, Kpad, outPad);
}

extern "C" void kernel_launch(void* const* d_in, const int* in_sizes, int n_in,
                              void* d_out, int out_size, void* d_ws, size_t ws_size,
                              hipStream_t stream) {
  (void)n_in; (void)out_size; (void)ws_size;

  const float* f_atoms  = (const float*)d_in[0];
  const float* f_bonds  = (const float*)d_in[1];
  const int*   a2b      = (const int*)d_in[2];
  const int*   b2a      = (const int*)d_in[3];
  const int*   b2revb   = (const int*)d_in[4];
  // d_in[5]=num_mols, d_in[6]=atoms_per_mol (device scalars; derived on host below)
  const float* W_i_atom = (const float*)d_in[7];
  const float* W_i_bond = (const float*)d_in[8];
  const float* W_h      = (const float*)d_in[9];    // [2,300,300]
  const float* lr_W     = (const float*)d_in[10];   // [300,900]
  const float* W_o_W    = (const float*)d_in[11];   // [300,600]
  const float* W_o_b    = (const float*)d_in[12];
  const float* gWih_f   = (const float*)d_in[13];
  const float* gWhh_f   = (const float*)d_in[14];
  const float* gbih_f   = (const float*)d_in[15];
  const float* gbhh_f   = (const float*)d_in[16];
  const float* gWih_b   = (const float*)d_in[17];
  const float* gWhh_b   = (const float*)d_in[18];
  const float* gbih_b   = (const float*)d_in[19];
  const float* gbhh_b   = (const float*)d_in[20];
  float* out = (float*)d_out;

  const int N1 = in_sizes[0] / AFD;     // N+1 atom rows (32769)
  const int B1 = in_sizes[1] / BFD;     // B+1 bond rows (131073)
  const int M  = (N1 - 1) / APM;        // molecules (512)

  // ---- workspace partition (256B aligned), with time-disjoint aliasing ----
  char* ws = (char*)d_ws;
  size_t off = 0;
  auto alloc = [&](size_t bytes) -> void* {
    void* p = ws + off;
    off += (bytes + 255) & ~(size_t)255;
    return p;
  };
  float* input_atom   = (float*)alloc((size_t)N1 * LD * 4);
  float* input_bond   = (float*)alloc((size_t)B1 * LD * 4);
  float* message_atom = (float*)alloc((size_t)N1 * LD * 4);
  float* message_bond = (float*)alloc((size_t)B1 * LD * 4);
  float* tmpbig       = (float*)alloc((size_t)B1 * LD * 4);  // fbonds_pad / bond tmp / node_in / grumsg
  float* node         = (float*)alloc((size_t)N1 * LD * 4);  // also fatoms_pad
  float* h_f  = (float*)alloc((size_t)M * LD * 4);
  float* h_b  = (float*)alloc((size_t)M * LD * 4);
  float* gh_f = (float*)alloc((size_t)M * GP * 4);
  float* gh_b = (float*)alloc((size_t)M * GP * 4);
  bf16* wpA   = (bf16*)alloc((size_t)320 * FPAD * 2);
  bf16* wpB   = (bf16*)alloc((size_t)320 * FPAD * 2);
  bf16* wph0  = (bf16*)alloc((size_t)320 * 320 * 2);
  bf16* wph1  = (bf16*)alloc((size_t)320 * 320 * 2);
  bf16* wplr  = (bf16*)alloc((size_t)320 * GP * 2);
  bf16* wpo   = (bf16*)alloc((size_t)320 * 608 * 2);
  bf16* wpihf = (bf16*)alloc((size_t)GP * 320 * 2);
  bf16* wphhf = (bf16*)alloc((size_t)GP * 320 * 2);
  bf16* wpihb = (bf16*)alloc((size_t)GP * 320 * 2);
  bf16* wphhb = (bf16*)alloc((size_t)GP * 320 * 2);
  // aliases (disjoint in time):
  float* fatoms_pad = node;           // used only before node exists
  float* fbonds_pad = tmpbig;         // used only before bond tmp
  float* gi_f    = input_bond;        // free after last bond GEMM
  float* gi_b    = message_bond;      // free after build_node_in
  float* node_in = tmpbig;            // used only for the readout GEMM
  float* grumsg  = tmpbig;            // written after node GEMM consumes node_in
  float* ah      = input_atom;        // atom_hiddens, after input_atom is done

  const int thrNH = N1 * HDIM, thrMH = M * HDIM;
  const int blkNH = (thrNH + 255) / 256, blkMH = (thrMH + 255) / 256;
  const long thrBL = (long)B1 * LD;
  const int  blkBL = (int)((thrBL + 255) / 256);

  // ---- 1. pack weights to bf16; repack features to aligned padded f32 ----
  launch_pack(W_i_atom, wpA, HDIM, AFD, stream);
  launch_pack(W_i_bond, wpB, HDIM, BFD, stream);
  launch_pack(W_h,               wph0, HDIM, HDIM, stream);
  launch_pack(W_h + HDIM * HDIM, wph1, HDIM, HDIM, stream);
  launch_pack(lr_W,  wplr, HDIM, G3,   stream);
  launch_pack(W_o_W, wpo,  HDIM, MSG2, stream);
  launch_pack(gWih_f, wpihf, G3, HDIM, stream);
  launch_pack(gWhh_f, wphhf, G3, HDIM, stream);
  launch_pack(gWih_b, wpihb, G3, HDIM, stream);
  launch_pack(gWhh_b, wphhb, G3, HDIM, stream);
  {
    long ta = (long)N1 * FPAD, tb = (long)B1 * FPAD;
    pack_act_kernel<<<(int)((ta + 255) / 256), 256, 0, stream>>>(f_atoms, fatoms_pad, N1, AFD, FPAD);
    pack_act_kernel<<<(int)((tb + 255) / 256), 256, 0, stream>>>(f_bonds, fbonds_pad, B1, BFD, FPAD);
  }

  // ---- 2. input projections (ReLU); aligned A => fully vectorized staging ----
  launch_gemm(fatoms_pad, FPAD, N1, FPAD, wpA, FPAD, nullptr, nullptr, 0,
              input_atom, LD, HDIM, LD, 0, 1, stream);
  launch_gemm(fbonds_pad, FPAD, B1, FPAD, wpB, FPAD, nullptr, nullptr, 0,
              input_bond, LD, HDIM, LD, 0, 1, stream);
  hipMemcpyAsync(message_atom, input_atom, (size_t)N1 * LD * 4,
                 hipMemcpyDeviceToDevice, stream);
  hipMemcpyAsync(message_bond, input_bond, (size_t)B1 * LD * 4,
                 hipMemcpyDeviceToDevice, stream);

  // ---- 3. message passing (depth-1 = 2). K=320: pads are zeroed ----
  const bf16* wph[2] = {wph0, wph1};
  for (int d = 0; d < 2; ++d) {
    agg_add_kernel<<<blkNH, 256, 0, stream>>>(message_bond, a2b, message_atom, N1);
    bond_tmp_kernel<<<blkBL, 256, 0, stream>>>(message_atom, message_bond, b2a, b2revb,
                                               tmpbig, B1);
    launch_gemm(tmpbig, LD, B1, LD, wph[d], 320, nullptr, input_bond, LD,
                message_bond, LD, HDIM, LD, 0, 1, stream);
  }

  // ---- 4. readout: node = concat(agg, message_atom, input_atom) @ lr_W^T ----
  build_node_in_kernel<<<blkNH, 256, 0, stream>>>(message_bond, a2b, message_atom,
                                                  input_atom, node_in, N1);
  launch_gemm(node_in, CATW, N1, CATW, wplr, GP, nullptr, nullptr, 0,
              node, LD, HDIM, LD, 0, 0, stream);

  // ---- 5. BatchGRU ----
  h0_kernel<<<(M * LD + 255) / 256, 256, 0, stream>>>(node, h_f, h_b, M);
  // input projections for all timesteps, both directions (ReLU on A = msg)
  launch_gemm(node + LD, LD, M * APM, LD, wpihf, 320, gbih_f, nullptr, 0,
              gi_f, GP, G3, GP, 1, 0, stream);
  launch_gemm(node + LD, LD, M * APM, LD, wpihb, 320, gbih_b, nullptr, 0,
              gi_b, GP, G3, GP, 1, 0, stream);
  row0_kernel<<<(HDIM + 255) / 256, 256, 0, stream>>>(node, grumsg);
  for (int t = 0; t < APM; ++t) {
    launch_gemm(h_f, LD, M, LD, wphhf, 320, gbhh_f, nullptr, 0,
                gh_f, GP, G3, GP, 0, 0, stream);
    gru_gate_kernel<<<blkMH, 256, 0, stream>>>(gi_f, gh_f, h_f, grumsg, M, t, 0);
    launch_gemm(h_b, LD, M, LD, wphhb, 320, gbhh_b, nullptr, 0,
                gh_b, GP, G3, GP, 0, 0, stream);
    gru_gate_kernel<<<blkMH, 256, 0, stream>>>(gi_b, gh_b, h_b, grumsg, M, t, 1);
  }

  // ---- 6. output head + per-molecule mean ----
  launch_gemm(grumsg, MSG2, N1, MSG2, wpo, 608, W_o_b, nullptr, 0,
              ah, LD, HDIM, LD, 0, 1, stream);
  mean_kernel<<<blkMH, 256, 0, stream>>>(ah, out, M);
}
